// SpectrumAttention_45449343926360
// MI455X (gfx1250) — matrix operand order repaired and checked by reading, not compiled, not run on hardware
//
#include <hip/hip_runtime.h>
#include <hip/hip_bf16.h>
#include <math.h>

// Problem constants (reference: B,C,N = 8,256,2048; H = C/8 = 32)
#define BB 8
#define CC 256
#define NN 2048
#define HH 32

typedef _Float16 h_t;
typedef __attribute__((ext_vector_type(8)))  _Float16 v8h;
typedef __attribute__((ext_vector_type(16))) _Float16 v16h;
typedef __attribute__((ext_vector_type(8)))  float    v8f;
typedef __attribute__((ext_vector_type(4)))  unsigned int u32x4;
typedef __attribute__((ext_vector_type(4)))  int i32x4;
typedef __attribute__((ext_vector_type(8)))  int i32x8;

static __device__ __forceinline__ v16h cat16(v8h a, v8h b) {
  return __builtin_shufflevector(a, b, 0,1,2,3,4,5,6,7,8,9,10,11,12,13,14,15);
}

static __device__ __forceinline__ v8f wmma16(v16h a, v16h b, v8f c) {
  return __builtin_amdgcn_wmma_f32_16x16x32_f16(
      /*neg_a=*/false, a, /*neg_b=*/false, b,
      /*c_mod=*/(short)0, c, /*reuse_a=*/false, /*reuse_b=*/false);
}

// ---------------------------------------------------------------------------
// TDM descriptor helpers (CDNA5 ISA ch.8). 2D tile, data_size = 2 bytes,
// LDS padding: pad_interval = 16 DWORDs (one 32-half row), pad_amount =
// 4 DWORDs -> LDS row stride = 20 DWORDs = 40 halves = 80 bytes.
// ---------------------------------------------------------------------------
#define LROW 40  // padded LDS row stride (halves) produced by TDM padding

static __device__ __forceinline__ u32x4 tdm_g0(unsigned int lds_addr,
                                               unsigned long long gaddr) {
  u32x4 g;
  g[0] = 1u;                                  // count = 1 (valid descriptor)
  g[1] = lds_addr;                            // LDS byte address of tile
  g[2] = (unsigned int)gaddr;                 // global tile address [31:0]
  g[3] = (unsigned int)((gaddr >> 32) & 0x01FFFFFFu) | (2u << 30); // type=2
  return g;
}

static __device__ __forceinline__ i32x8 tdm_g1(unsigned int dim0,
                                               unsigned int dim1,
                                               unsigned int tile0,
                                               unsigned int tile1,
                                               unsigned long long stride0) {
  i32x8 g = {};
  // data_size=2B (1<<16), pad_enable (1<<20), pad_interval=3 -> 16 DWORDs,
  // pad_amount=3 -> 4 DWORDs
  g[0] = (int)((1u << 16) | (1u << 20) | (3u << 22) | (3u << 25));
  g[1] = (int)((dim0 & 0xFFFFu) << 16);                      // tensor_dim0 lo
  g[2] = (int)((dim0 >> 16) | ((dim1 & 0xFFFFu) << 16));     // dim0 hi|dim1 lo
  g[3] = (int)((dim1 >> 16) | (tile0 << 16));                // dim1 hi|tile0
  g[4] = (int)(tile1 & 0xFFFFu);                             // tile1, tile2=0
  g[5] = (int)(unsigned int)(stride0 & 0xFFFFFFFFu);         // dim0_stride lo
  g[6] = (int)(unsigned int)((stride0 >> 32) & 0xFFFFu);     // dim0_stride hi
  g[7] = 0;
  return g;
}

static __device__ __forceinline__ void tdm_load(u32x4 g0, i32x8 g1) {
  i32x4 z4 = {};
#if __clang_major__ >= 23
  i32x8 z8 = {};
  __builtin_amdgcn_tensor_load_to_lds(g0, g1, z4, z4, z8, 0);
#else
  __builtin_amdgcn_tensor_load_to_lds(g0, g1, z4, z4, 0);
#endif
}

// ---------------------------------------------------------------------------
// Kernel 0a: x [B][C][N] f32  ->  xth [B][N][C] f16   (transpose via LDS tile)
// ---------------------------------------------------------------------------
__global__ __launch_bounds__(256) void k_xpose(const float* __restrict__ x,
                                               h_t* __restrict__ xth) {
  __shared__ h_t tile[32 * 33];
  const int bid  = blockIdx.x;
  const int nblk = bid % (NN / 32);
  const int cblk = (bid / (NN / 32)) % (CC / 32);
  const int b    = bid / ((NN / 32) * (CC / 32));
  const int t0   = threadIdx.x & 31;
  const int t1   = threadIdx.x >> 5;
#pragma unroll
  for (int p = 0; p < 4; ++p) {
    const int c = cblk * 32 + t1 + p * 8;
    tile[(t1 + p * 8) * 33 + t0] =
        (h_t)x[((size_t)b * CC + c) * NN + nblk * 32 + t0];
  }
  __syncthreads();
#pragma unroll
  for (int p = 0; p < 4; ++p) {
    const int n = nblk * 32 + t1 + p * 8;
    xth[((size_t)b * NN + n) * CC + cblk * 32 + t0] =
        tile[t0 * 33 + t1 + p * 8];
  }
}

// ---------------------------------------------------------------------------
// Kernel 0b: weights f32 -> f16 (kept [out][in] row-major)
// ---------------------------------------------------------------------------
__global__ __launch_bounds__(256) void k_wconv(const float* __restrict__ wq,
                                               const float* __restrict__ wk,
                                               const float* __restrict__ wv,
                                               h_t* __restrict__ wqh,
                                               h_t* __restrict__ wkh,
                                               h_t* __restrict__ wvh) {
  const int i = blockIdx.x * 256 + threadIdx.x;
  if (i < HH * CC) { wqh[i] = (h_t)wq[i]; wkh[i] = (h_t)wk[i]; }
  if (i < CC * CC) { wvh[i] = (h_t)wv[i]; }
}

// ---------------------------------------------------------------------------
// One 16x16 output tile of  out[n, o] = sum_c xth[n][c] * w[o][c]   (K = 256)
// ---------------------------------------------------------------------------
static __device__ __forceinline__ v8f gemm_tile(const h_t* __restrict__ xrow,
                                                const h_t* __restrict__ w,
                                                int o0, int hi, int l16) {
  v8f acc = {};
#pragma unroll
  for (int c0 = 0; c0 < CC; c0 += 32) {
    const h_t* ap = xrow + c0 + (hi ? 8 : 0);
    v16h A = cat16(*(const v8h*)ap, *(const v8h*)(ap + 16));
    const h_t* bp = w + (size_t)(o0 + l16) * CC + c0 + (hi ? 8 : 0);
    v16h Bm = cat16(*(const v8h*)bp, *(const v8h*)(bp + 16));
    acc = wmma16(A, Bm, acc);
  }
  return acc;
}

// ---------------------------------------------------------------------------
// Kernel 1: projections.  One wave = 16 rows of n.
//   qh[b][n][h], kh[b][n][h]  (H=32 contiguous => one WMMA K-step later)
//   vh[b][c][n]               (row-major over n => contiguous B operand later)
// ---------------------------------------------------------------------------
__global__ __launch_bounds__(128) void k_proj(const h_t* __restrict__ xth,
                                              const h_t* __restrict__ wqh,
                                              const h_t* __restrict__ wkh,
                                              const h_t* __restrict__ wvh,
                                              const float* __restrict__ bq,
                                              const float* __restrict__ bk,
                                              const float* __restrict__ bv,
                                              h_t* __restrict__ qh,
                                              h_t* __restrict__ kh,
                                              h_t* __restrict__ vh) {
  const int lane = threadIdx.x & 31;
  const int wave = threadIdx.x >> 5;
  const int hi   = (lane >> 4) & 1;
  const int l16  = lane & 15;
  const int b    = blockIdx.x / (NN / 64);
  const int n0   = (blockIdx.x % (NN / 64)) * 64 + wave * 16;

  const h_t* xrow = xth + ((size_t)b * NN + n0 + l16) * CC;

#pragma unroll
  for (int o0 = 0; o0 < HH; o0 += 16) {
    v8f aq = gemm_tile(xrow, wqh, o0, hi, l16);
    v8f ak = gemm_tile(xrow, wkh, o0, hi, l16);
    const float bbq = bq[o0 + l16];
    const float bbk = bk[o0 + l16];
#pragma unroll
    for (int r = 0; r < 8; ++r) {
      const size_t row = (size_t)b * NN + n0 + r + (hi ? 8 : 0);
      qh[row * HH + o0 + l16] = (h_t)(aq[r] + bbq);
      kh[row * HH + o0 + l16] = (h_t)(ak[r] + bbk);
    }
  }
#pragma unroll
  for (int o0 = 0; o0 < CC; o0 += 16) {
    v8f av = gemm_tile(xrow, wvh, o0, hi, l16);
    const float bbv = bv[o0 + l16];
    v8h st;
#pragma unroll
    for (int r = 0; r < 8; ++r) st[r] = (h_t)(av[r] + bbv);
    *(v8h*)(vh + ((size_t)b * CC + o0 + l16) * NN + n0 + (hi ? 8 : 0)) = st;
  }
}

// ---------------------------------------------------------------------------
// Kernel 2: fused flash attention + PV + residual, TDM-fed.
// 4 waves / block share double-buffered K (32x32) and V (256x32) f16 tiles
// staged into LDS by the Tensor Data Mover (wave 0 issues, TENSORcnt-waited,
// s_barrier published).  Each wave owns a 16-row query slab with the full
// 16x256 O accumulator (16 WMMA D tiles) in registers.
// ---------------------------------------------------------------------------
#define PSTR 40  // padded LDS row stride for the P relayout buffer (halves)

__global__ __launch_bounds__(128) void k_attn(const h_t* __restrict__ qh,
                                              const h_t* __restrict__ kh,
                                              const h_t* __restrict__ vh,
                                              const float* __restrict__ x,
                                              float* __restrict__ out) {
  __shared__ __align__(128) h_t kbuf[2][32 * LROW];
  __shared__ __align__(128) h_t vbuf[2][CC * LROW];
  __shared__ __align__(128) h_t pbuf_all[4][16 * PSTR];

  const int lane = threadIdx.x & 31;
  const int wave = threadIdx.x >> 5;
  const int hi   = (lane >> 4) & 1;
  const int l16  = lane & 15;
  const int b    = blockIdx.x / (NN / 64);
  const int i0   = (blockIdx.x % (NN / 64)) * 64 + wave * 16;
  h_t* pbuf = &pbuf_all[wave][0];

  const h_t* khb = kh + (size_t)b * NN * HH;   // K rows [n][h]
  const h_t* vhb = vh + (size_t)b * CC * NN;   // V rows [c][n]

  // Uniform group-1 descriptors (tile shape/strides constant over the loop)
  const i32x8 g1k = tdm_g1(/*dim0=*/HH, /*dim1=*/NN, /*tile0=*/32,
                           /*tile1=*/32, /*stride0=*/HH);
  const i32x8 g1v = tdm_g1(/*dim0=*/NN, /*dim1=*/CC, /*tile0=*/32,
                           /*tile1=*/CC, /*stride0=*/NN);

  // Prologue: stage tile j0 = 0 into buffer 0.
  if (wave == 0) {
    tdm_load(tdm_g0((unsigned int)(uintptr_t)&kbuf[0][0],
                    (unsigned long long)(uintptr_t)khb), g1k);
    tdm_load(tdm_g0((unsigned int)(uintptr_t)&vbuf[0][0],
                    (unsigned long long)(uintptr_t)vhb), g1v);
    __builtin_amdgcn_s_wait_tensorcnt((short)0);
  }
  __syncthreads();

  // A operand: q slab, loaded once (K = H = 32)
  const h_t* qrow = qh + ((size_t)b * NN + (i0 + l16)) * HH;
  const v16h Aq = cat16(*(const v8h*)(qrow + (hi ? 8 : 0)),
                        *(const v8h*)(qrow + 16 + (hi ? 8 : 0)));

  float mrow[8], lrow[8];
  v8f O[16];
#pragma unroll
  for (int r = 0; r < 8; ++r) { mrow[r] = -3.0e38f; lrow[r] = 0.0f; }
#pragma unroll
  for (int ct = 0; ct < 16; ++ct) O[ct] = (v8f){};

  for (int j0 = 0; j0 < NN; j0 += 32) {
    const int cur = (j0 >> 5) & 1;
    // Kick off the DMA for the next tile while computing on this one.
    if (wave == 0 && j0 + 32 < NN) {
      tdm_load(tdm_g0((unsigned int)(uintptr_t)&kbuf[cur ^ 1][0],
                      (unsigned long long)(uintptr_t)(khb +
                          (size_t)(j0 + 32) * HH)), g1k);
      tdm_load(tdm_g0((unsigned int)(uintptr_t)&vbuf[cur ^ 1][0],
                      (unsigned long long)(uintptr_t)(vhb + j0 + 32)), g1v);
    }

    // S tiles from LDS K tile: B operand = halves 8h..8h+7 and 16+8h..16+8h+7 of row j
    const h_t* k0 = &kbuf[cur][0] + l16 * LROW + (hi ? 8 : 0);
    const h_t* k1 = &kbuf[cur][0] + (16 + l16) * LROW + (hi ? 8 : 0);
    v16h B0 = cat16(*(const v8h*)k0, *(const v8h*)(k0 + 16));
    v16h B1 = cat16(*(const v8h*)k1, *(const v8h*)(k1 + 16));
    v8f z = {};
    v8f S0 = wmma16(Aq, B0, z);
    v8f S1 = wmma16(Aq, B1, z);

    // Online softmax. D-layout: element (M = r + 8*hi, N = l16) -> each
    // half-wave holds a disjoint set of 8 rows, reductions stay in-half.
#pragma unroll
    for (int r = 0; r < 8; ++r) {
      float s0 = S0[r], s1 = S1[r];
      float mx = fmaxf(s0, s1);
      mx = fmaxf(mx, __shfl_xor(mx, 1, 16));
      mx = fmaxf(mx, __shfl_xor(mx, 2, 16));
      mx = fmaxf(mx, __shfl_xor(mx, 4, 16));
      mx = fmaxf(mx, __shfl_xor(mx, 8, 16));
      const float mn   = fmaxf(mrow[r], mx);
      const float corr = __expf(mrow[r] - mn);
      const float p0 = __expf(s0 - mn);
      const float p1 = __expf(s1 - mn);
      float rs = p0 + p1;
      rs += __shfl_xor(rs, 1, 16);
      rs += __shfl_xor(rs, 2, 16);
      rs += __shfl_xor(rs, 4, 16);
      rs += __shfl_xor(rs, 8, 16);
      lrow[r] = lrow[r] * corr + rs;
      mrow[r] = mn;
#pragma unroll
      for (int ct = 0; ct < 16; ++ct) O[ct][r] *= corr;
      const int prow = r + (hi ? 8 : 0);
      pbuf[prow * PSTR + l16]      = (h_t)p0;
      pbuf[prow * PSTR + 16 + l16] = (h_t)p1;
    }
    // same-wave LDS RAW: DS ops are in-order per wave; wait + sched fence
    asm volatile("s_wait_dscnt 0x0" ::: "memory");

    // P as A operand (K = j, 32 wide)
    const h_t* pr = pbuf + l16 * PSTR;
    v16h Ap = cat16(*(const v8h*)(pr + (hi ? 8 : 0)),
                    *(const v8h*)(pr + 16 + (hi ? 8 : 0)));
    // O += P * V^T from LDS V tile
#pragma unroll
    for (int ct = 0; ct < 16; ++ct) {
      const h_t* vp = &vbuf[cur][0] + (ct * 16 + l16) * LROW + (hi ? 8 : 0);
      v16h Bv = cat16(*(const v8h*)vp, *(const v8h*)(vp + 16));
      O[ct] = wmma16(Ap, Bv, O[ct]);
    }

    // Publish the next buffer: issuing wave drains TENSORcnt, then barrier.
    if (wave == 0) __builtin_amdgcn_s_wait_tensorcnt((short)0);
    __syncthreads();
  }

  // Epilogue: out[b][c][i] = O/l + x  (per lane: 8 contiguous i's per tile)
#pragma unroll
  for (int ct = 0; ct < 16; ++ct) {
    const int c = ct * 16 + l16;
    const size_t base = ((size_t)b * CC + c) * NN + i0 + (hi ? 8 : 0);
#pragma unroll
    for (int r = 0; r < 8; ++r) {
      out[base + r] = O[ct][r] / lrow[r] + x[base + r];
    }
  }
}

// ---------------------------------------------------------------------------
extern "C" void kernel_launch(void* const* d_in, const int* in_sizes, int n_in,
                              void* d_out, int out_size, void* d_ws,
                              size_t ws_size, hipStream_t stream) {
  (void)in_sizes; (void)n_in; (void)out_size; (void)ws_size;
  const float* x  = (const float*)d_in[0];
  const float* wq = (const float*)d_in[1];
  const float* bq = (const float*)d_in[2];
  const float* wk = (const float*)d_in[3];
  const float* bk = (const float*)d_in[4];
  const float* wv = (const float*)d_in[5];
  const float* bv = (const float*)d_in[6];
  float* out = (float*)d_out;

  char* ws = (char*)d_ws;
  size_t off = 0;
  h_t* xth = (h_t*)(ws + off); off += (size_t)BB * NN * CC * sizeof(h_t);
  h_t* wqh = (h_t*)(ws + off); off += (size_t)HH * CC * sizeof(h_t);
  h_t* wkh = (h_t*)(ws + off); off += (size_t)HH * CC * sizeof(h_t);
  h_t* wvh = (h_t*)(ws + off); off += (size_t)CC * CC * sizeof(h_t);
  h_t* qh  = (h_t*)(ws + off); off += (size_t)BB * NN * HH * sizeof(h_t);
  h_t* kh  = (h_t*)(ws + off); off += (size_t)BB * NN * HH * sizeof(h_t);
  h_t* vh  = (h_t*)(ws + off); off += (size_t)BB * CC * NN * sizeof(h_t);

  k_xpose<<<BB * (CC / 32) * (NN / 32), 256, 0, stream>>>(x, xth);
  k_wconv<<<(CC * CC + 255) / 256, 256, 0, stream>>>(wq, wk, wv, wqh, wkh, wvh);
  k_proj<<<BB * (NN / 64), 128, 0, stream>>>(xth, wqh, wkh, wvh, bq, bk, bv,
                                             qh, kh, vh);
  k_attn<<<BB * (NN / 64), 128, 0, stream>>>(qh, kh, vh, x, out);
}
